// Retina2_18339510354527
// MI455X (gfx1250) — compile-verified, matrix operand order checked by reading, not run on hardware
//
#include <hip/hip_runtime.h>
#include <math.h>

#define B_ 16
#define A_ 32526
#define G_ 64
#define C_ 3
#define NBA (B_*A_)

typedef float    v2f  __attribute__((ext_vector_type(2)));
typedef float    v8f  __attribute__((ext_vector_type(8)));
typedef _Float16 v16h __attribute__((ext_vector_type(16)));

#if __has_builtin(__builtin_amdgcn_wmma_f32_16x16x4_f32)
#define RETINA_WMMA_F32 1
#endif

__device__ __forceinline__ float sigmoid_clip(float x) {
    float p = 1.0f / (1.0f + expf(-x));
    return fminf(fmaxf(p, 1e-6f), 1.0f - 1e-6f);
}

// ---------------------------------------------------------------- K0: init
__global__ void k_init(unsigned long long* packed, int* npos, float* sums) {
    int i = blockIdx.x * 256 + threadIdx.x;
    if (i < B_ * G_) packed[i] = 0ULL;
    if (i < B_)      npos[i]   = 0;
    if (i < B_ * 4)  sums[i]   = 0.0f;
}

// ------------------------------------------------- K1: IoU + matching maxes
__global__ void k_iou(const float* __restrict__ gboxes,
                      const float* __restrict__ anchors,
                      float* __restrict__ iou_anc, int* __restrict__ idx0,
                      unsigned long long* __restrict__ packed) {
    __shared__ float sgb[G_ * 4];
    __shared__ unsigned long long best[G_];
    const int b = blockIdx.y;
    const int t = threadIdx.x;
    sgb[t] = gboxes[b * G_ * 4 + t];          // 256 floats == G_*4
    if (t < G_) best[t] = 0ULL;
    __syncthreads();

    const int a = blockIdx.x * 256 + t;
    if (a < A_) {
        float cx = anchors[a * 4 + 0], cy = anchors[a * 4 + 1];
        float aw = anchors[a * 4 + 2], ah = anchors[a * 4 + 3];
        float al = cx - aw * 0.5f, at = cy - ah * 0.5f;
        float ar = cx + aw * 0.5f, ab = cy + ah * 0.5f;
        float area_a = (ar - al) * (ab - at);
        float bi = -1.0f; int bg = 0;
        for (int g = 0; g < G_; ++g) {
            float gl = sgb[g * 4 + 0], gt = sgb[g * 4 + 1];
            float gr = sgb[g * 4 + 2], gb = sgb[g * 4 + 3];
            float lt0 = fmaxf(al, gl), lt1 = fmaxf(at, gt);
            float rb0 = fminf(ar, gr), rb1 = fminf(ab, gb);
            float w = fmaxf(rb0 - lt0, 0.0f), h = fmaxf(rb1 - lt1, 0.0f);
            float inter  = w * h;
            float area_g = (gr - gl) * (gb - gt);
            float iou = inter / (area_a + area_g - inter + 1e-6f);
            if (iou > bi) { bi = iou; bg = g; }     // first-occurrence argmax
            unsigned ib = __float_as_uint(iou);      // iou >= 0, order-preserving
            unsigned long long p =
                ((unsigned long long)ib << 32) | (unsigned)(~(unsigned)a);
            atomicMax(&best[g], p);                  // ds_max_u64
        }
        iou_anc[b * A_ + a] = bi;
        idx0[b * A_ + a]    = bg;
    }
    __syncthreads();
    if (t < G_) atomicMax(&packed[b * G_ + t], best[t]);  // global_atomic_max_u64
}

// --------------------------------------------------------- K2: decode bestA
__global__ void k_decode(const unsigned long long* __restrict__ packed,
                         int* __restrict__ bestA) {
    int i = blockIdx.x * 256 + threadIdx.x;
    if (i < B_ * G_)
        bestA[i] = (int)(~(unsigned)(packed[i] & 0xFFFFFFFFULL));
}

// ------------------------------------------- K3: per-anchor loss components
__global__ void k_loss(const float* __restrict__ ptxywh,
                       const float* __restrict__ pcat,
                       const float* __restrict__ gboxes,
                       const float* __restrict__ anchors,
                       const int*   __restrict__ glabels,
                       const float* __restrict__ iou_anc,
                       const int*   __restrict__ idx0,
                       const int*   __restrict__ bestA,
                       float* __restrict__ lcp, float* __restrict__ lcl,
                       float* __restrict__ lbx, float* __restrict__ lng,
                       int* __restrict__ npos) {
    __shared__ float sgb[G_ * 4];
    __shared__ int   slab[G_];
    __shared__ int   sba[G_];
    __shared__ int   scnt;
    const int b = blockIdx.y;
    const int t = threadIdx.x;
    sgb[t] = gboxes[b * G_ * 4 + t];
    if (t < G_) { slab[t] = glabels[b * G_ + t]; sba[t] = bestA[b * G_ + t]; }
    if (t == 0) scnt = 0;
    __syncthreads();

    const int a = blockIdx.x * 256 + t;
    if (a < A_) {
        const int   base = b * A_ + a;
        float iou = iou_anc[base];
        int   idx = idx0[base];
        float gconf = (iou >= 0.5f) ? 1.0f : ((iou >= 0.4f) ? -1.0f : 0.0f);
        for (int g = 0; g < G_; ++g)                 // ascending => last-wins
            if (sba[g] == a) { gconf = 1.0f; idx = g; }
        bool pos = (gconf > 0.0f);

        float p  = sigmoid_clip(pcat[base * 4 + 0]);
        float lv = pos ? -logf(p) : -logf(1.0f - p); // ign branch unused downstream
        lcp[base] = pos ? lv : 0.0f;
        lng[base] = (gconf == 0.0f) ? lv : 0.0f;

        float cl = 0.0f, bx = 0.0f;
        if (pos) {
            int lbl = slab[idx] - 1;
            for (int c = 0; c < C_; ++c) {
                float pc = sigmoid_clip(pcat[base * 4 + 1 + c]);
                cl += (c == lbl) ? -logf(pc) : -logf(1.0f - pc);
            }
            float gl = sgb[idx * 4 + 0], gt = sgb[idx * 4 + 1];
            float gr = sgb[idx * 4 + 2], gb = sgb[idx * 4 + 3];
            float gx = (gl + gr) * 0.5f, gy = (gt + gb) * 0.5f;
            float gw = fmaxf(gr - gl, 1e-6f), gh = fmaxf(gb - gt, 1e-6f);
            float acx = anchors[a * 4 + 0], acy = anchors[a * 4 + 1];
            float aw  = anchors[a * 4 + 2], ah  = anchors[a * 4 + 3];
            float tt[4];
            tt[0] = (gx - acx) / (aw * 0.1f);
            tt[1] = (gy - acy) / (ah * 0.1f);
            tt[2] = logf(gw / aw) / 0.2f;
            tt[3] = logf(gh / ah) / 0.2f;
            for (int j = 0; j < 4; ++j) {
                float d = ptxywh[base * 4 + j] - tt[j];
                float ad = fabsf(d);
                bx += (ad < 1.0f) ? 0.5f * d * d : ad - 0.5f;
            }
            atomicAdd(&scnt, 1);
        }
        lcl[base] = cl;
        lbx[base] = bx;
    }
    __syncthreads();
    if (t == 0 && scnt) atomicAdd(&npos[b], scnt);
}

// ------------------------------- K4: WMMA ones-trick per-batch loss reduction
__global__ void k_reduce(const float* __restrict__ lcp,
                         const float* __restrict__ lcl,
                         const float* __restrict__ lbx,
                         float* __restrict__ sums) {
    const int b = blockIdx.x;
    const int t = threadIdx.x;
    __shared__ float acc[3];
    if (t < 3) acc[t] = 0.0f;
    __syncthreads();

#if defined(RETINA_WMMA_F32)
    v2f ones; ones[0] = 1.0f; ones[1] = 1.0f;          // A = ones(16x4), exact
    v8f c0 = {}, c1 = {}, c2 = {};
    for (int base = 0; base < A_; base += 256) {        // uniform trip count
        int a = base + t;
        float v0 = 0.0f, v1 = 0.0f, v2 = 0.0f;
        if (a < A_) {
            v0 = lcp[b * A_ + a]; v1 = lcl[b * A_ + a]; v2 = lbx[b * A_ + a];
        }
        v2f B0; B0[0] = v0; B0[1] = 0.0f;
        v2f B1; B1[0] = v1; B1[1] = 0.0f;
        v2f B2; B2[0] = v2; B2[1] = 0.0f;
        // D = ones x B + C : every D column = sum of 4 B slots; 16 columns
        // cover all 64 per-wave B slots exactly once.
        c0 = __builtin_amdgcn_wmma_f32_16x16x4_f32(false, ones, false, B0, (short)0, c0, false, false);
        c1 = __builtin_amdgcn_wmma_f32_16x16x4_f32(false, ones, false, B1, (short)0, c1, false, false);
        c2 = __builtin_amdgcn_wmma_f32_16x16x4_f32(false, ones, false, B2, (short)0, c2, false, false);
    }
#else
    // Fallback: f16 WMMA with exact hi/lo split (f32 accumulate in C/D).
    v16h ones;
    for (int i = 0; i < 16; ++i) ones[i] = (_Float16)1.0f;
    v8f c0 = {}, c1 = {}, c2 = {};
    for (int base = 0; base < A_; base += 256) {
        int a = base + t;
        float v0 = 0.0f, v1 = 0.0f, v2 = 0.0f;
        if (a < A_) {
            v0 = lcp[b * A_ + a]; v1 = lcl[b * A_ + a]; v2 = lbx[b * A_ + a];
        }
        v16h B0 = {}, B1 = {}, B2 = {};
        _Float16 h;
        h = (_Float16)v0; B0[0] = h; B0[1] = (_Float16)(v0 - (float)h);
        h = (_Float16)v1; B1[0] = h; B1[1] = (_Float16)(v1 - (float)h);
        h = (_Float16)v2; B2[0] = h; B2[1] = (_Float16)(v2 - (float)h);
        c0 = __builtin_amdgcn_wmma_f32_16x16x32_f16(false, ones, false, B0, (short)0, c0, false, false);
        c1 = __builtin_amdgcn_wmma_f32_16x16x32_f16(false, ones, false, B1, (short)0, c1, false, false);
        c2 = __builtin_amdgcn_wmma_f32_16x16x32_f16(false, ones, false, B2, (short)0, c2, false, false);
    }
#endif
    // D VGPR0: lanes 0..15 hold the 16 column sums (row M=0). Sum them.
    if ((t & 31) < 16) {
        atomicAdd(&acc[0], c0[0]);   // ds_add_f32
        atomicAdd(&acc[1], c1[0]);
        atomicAdd(&acc[2], c2[0]);
    }
    __syncthreads();
    if (t < 3) sums[b * 4 + t] = acc[t];
}

// ------------------------ K5: exact top-K hard-negative sum via radix select
__global__ void k_topk(const float* __restrict__ lng,
                       const int* __restrict__ npos,
                       float* __restrict__ sums) {
    const int b = blockIdx.x;
    const int t = threadIdx.x;
    __shared__ unsigned hist[256];
    __shared__ unsigned sh_pfx;
    __shared__ int      sh_rem;
    __shared__ float    facc;

    int K = npos[b] * 3;
    if (K > A_) K = A_;
    if (t == 0) { sh_pfx = 0u; sh_rem = K; facc = 0.0f; }
    __syncthreads();

    for (int pass = 0; pass < 4; ++pass) {
        const int shift = 24 - pass * 8;
        hist[t] = 0u;
        __syncthreads();
        const unsigned pfx   = sh_pfx;
        const unsigned hmask = (pass == 0) ? 0u : (0xFFFFFFFFu << (shift + 8));
        for (int base = 0; base < A_; base += 256) {
            int a = base + t;
            if (a < A_) {
                unsigned bits = __float_as_uint(lng[b * A_ + a]);
                if ((bits & hmask) == pfx)
                    atomicAdd(&hist[(bits >> shift) & 0xFFu], 1u);
            }
        }
        __syncthreads();
        if (t == 0) {
            int rem = sh_rem;
            int dig;
            for (dig = 255; dig > 0; --dig) {
                int h = (int)hist[dig];
                if (h >= rem) break;
                rem -= h;
            }
            sh_rem = rem;
            sh_pfx = pfx | ((unsigned)dig << shift);
        }
        __syncthreads();
    }
    const unsigned T = sh_pfx;
    const int remT   = sh_rem;

    float local = 0.0f;
    for (int base = 0; base < A_; base += 256) {
        int a = base + t;
        if (a < A_) {
            float v = lng[b * A_ + a];
            if (__float_as_uint(v) > T) local += v;
        }
    }
    atomicAdd(&facc, local);
    __syncthreads();
    // Tied boundary values are bit-identical => stable-sort selection of
    // `remT` of them contributes exactly remT * float(T).
    if (t == 0) sums[b * 4 + 3] = facc + (float)remT * __uint_as_float(T);
}

// ---------------------------------------------------------------- K6: final
__global__ void k_final(const float* __restrict__ sums,
                        const int* __restrict__ npos,
                        float* __restrict__ out) {
    if (blockIdx.x == 0 && threadIdx.x == 0) {
        float tot = 0.0f;
        for (int b = 0; b < B_; ++b) {
            float np = fmaxf((float)npos[b], 0.0009765625f);   // FP16_EPS clip
            tot += (sums[b * 4 + 0] + sums[b * 4 + 1] +
                    sums[b * 4 + 2] + sums[b * 4 + 3]) / np;
        }
        out[0] = tot / (float)B_;
    }
}

extern "C" void kernel_launch(void* const* d_in, const int* in_sizes, int n_in,
                              void* d_out, int out_size, void* d_ws, size_t ws_size,
                              hipStream_t stream) {
    (void)in_sizes; (void)n_in; (void)out_size; (void)ws_size;
    const float* ptxywh  = (const float*)d_in[0];
    const float* pcat    = (const float*)d_in[1];
    const float* gboxes  = (const float*)d_in[2];
    const float* anchors = (const float*)d_in[3];
    const int*   glabels = (const int*)d_in[4];
    float* out = (float*)d_out;

    const size_t S = (size_t)NBA * sizeof(float);   // 2,081,664 B per array
    char* w = (char*)d_ws;
    unsigned long long* packed = (unsigned long long*)(w);        // 8 KiB
    int*   bestA = (int*)(w + 8192);                              // 4 KiB
    int*   npos  = (int*)(w + 8192 + 4096);                       // 64 B
    float* sums  = (float*)(w + 8192 + 4096 + 64);                // 256 B
    char*  arrs  = w + 16384;
    float* iou   = (float*)(arrs + 0 * S);
    int*   idx0  = (int*)  (arrs + 1 * S);
    float* lcp   = (float*)(arrs + 2 * S);
    float* lcl   = (float*)(arrs + 3 * S);
    float* lbx   = (float*)(arrs + 4 * S);
    float* lng   = (float*)(arrs + 5 * S);

    const int ABLK = (A_ + 255) / 256;   // 128
    k_init  <<<4, 256, 0, stream>>>(packed, npos, sums);
    k_iou   <<<dim3(ABLK, B_), 256, 0, stream>>>(gboxes, anchors, iou, idx0, packed);
    k_decode<<<4, 256, 0, stream>>>(packed, bestA);
    k_loss  <<<dim3(ABLK, B_), 256, 0, stream>>>(ptxywh, pcat, gboxes, anchors,
                                                 glabels, iou, idx0, bestA,
                                                 lcp, lcl, lbx, lng, npos);
    k_reduce<<<B_, 256, 0, stream>>>(lcp, lcl, lbx, sums);
    k_topk  <<<B_, 256, 0, stream>>>(lng, npos, sums);
    k_final <<<1, 32, 0, stream>>>(sums, npos, out);
}
